// Block_30176440222067
// MI455X (gfx1250) — matrix operand order repaired and checked by reading, not compiled, not run on hardware
//
#include <hip/hip_runtime.h>

typedef unsigned short ushort_t;
typedef unsigned int   uint32;

typedef __attribute__((ext_vector_type(16))) __bf16 v16bf;
typedef __attribute__((ext_vector_type(8)))  float  v8f;

union Frag {
  v16bf  v;
  uint32 u[8];
};

__device__ inline ushort_t f2bf(float f) {
  uint32 x = __float_as_uint(f);
  uint32 r = (x + 0x7FFFu + ((x >> 16) & 1u)) >> 16;  // round-to-nearest-even
  return (ushort_t)r;
}

__device__ inline v8f wmma_bf16(const Frag& a, const Frag& b, v8f c) {
  // D = A(16x32 bf16) * B(32x16 bf16) + C(16x16 f32)
  return __builtin_amdgcn_wmma_f32_16x16x32_bf16(
      /*neg_a=*/false, a.v, /*neg_b=*/false, b.v,
      /*c_mod=*/(short)0, c, /*reuse_a=*/false, /*reuse_b=*/false);
}

// ---------------------------------------------------------------------------
// LayerNorm over rows of 1024, fp32 in -> bf16 out. One block per row.
// ---------------------------------------------------------------------------
__global__ __launch_bounds__(256) void ln_kernel(const float* __restrict__ X,
                                                 const float* __restrict__ w,
                                                 const float* __restrict__ b,
                                                 ushort_t* __restrict__ out) {
  const int D = 1024;
  int row = blockIdx.x;
  const float* xr = X + (size_t)row * D;
  int t = threadIdx.x;
  float s = 0.f, sq = 0.f;
  float xs[4];
  #pragma unroll
  for (int j = 0; j < 4; ++j) {
    float v = xr[t + 256 * j];
    xs[j] = v;
    s += v;
    sq += v * v;
  }
  #pragma unroll
  for (int o = 1; o < 32; o <<= 1) {
    s  += __shfl_xor(s, o, 32);
    sq += __shfl_xor(sq, o, 32);
  }
  __shared__ float ssum[8], ssq[8];
  int wid = t >> 5;
  if ((t & 31) == 0) { ssum[wid] = s; ssq[wid] = sq; }
  __syncthreads();
  float ts = 0.f, tq = 0.f;
  #pragma unroll
  for (int i = 0; i < 8; ++i) { ts += ssum[i]; tq += ssq[i]; }
  float mean = ts * (1.0f / D);
  float var  = tq * (1.0f / D) - mean * mean;
  float rstd = rsqrtf(var + 1e-5f);
  #pragma unroll
  for (int j = 0; j < 4; ++j) {
    int i = t + 256 * j;
    out[(size_t)row * D + i] = f2bf((xs[j] - mean) * rstd * w[i] + b[i]);
  }
}

// ---------------------------------------------------------------------------
// GEMM: C[M,N] = A[M,K](bf16) * W[K,N](fp32, cast to bf16) + bias, epilogues:
//   EPI=0: store bf16      EPI=1: exact GELU, store bf16
//   EPI=2: + resid (fp32), store fp32
// Block tile 128(M) x 128(N), BK=32. 256 threads = 8 waves.
// Wave tile 32(M) x 64(N) = 2x4 WMMA subtiles -> 8 WMMAs per K-step.
// ---------------------------------------------------------------------------
template <int EPI>
__global__ __launch_bounds__(256) void gemm_bf16_kernel(
    const ushort_t* __restrict__ A, const float* __restrict__ W,
    const float* __restrict__ bias, const float* __restrict__ resid,
    ushort_t* __restrict__ outb, float* __restrict__ outf,
    int M, int N, int K) {
  __shared__ ushort_t As[128][34];  // [m][k] row-major, padded
  __shared__ ushort_t Bs[128][34];  // [n][k] (transposed), padded

  int t = threadIdx.x;
  int l = t & 31, w = t >> 5;
  int lm = l & 15, hx = l >> 4;
  int wm = w & 3, wn = w >> 2;      // 4 waves along M, 2 along N
  int m0 = blockIdx.y * 128;
  int n0 = blockIdx.x * 128;

  v8f acc[2][4] = {};

  int arow = t >> 1, akoff = (t & 1) * 16;  // 128 rows x 32 k, 16 elems/thread
  int bnn = t & 127, bkk0 = (t >> 7) * 16;  // 128 cols x 32 k, 16 elems/thread

  const ushort_t* aptr = A + (size_t)(m0 + arow) * K + akoff;
  const float*    wptr = W + n0 + bnn;

  for (int k0 = 0; k0 < K; k0 += 32) {
    const uint32* ap = (const uint32*)(aptr + k0);
    #pragma unroll
    for (int i = 0; i < 8; ++i)
      *(uint32*)&As[arow][akoff + 2 * i] = ap[i];
    #pragma unroll
    for (int i = 0; i < 16; ++i) {
      int kk = bkk0 + i;
      Bs[bnn][kk] = f2bf(wptr[(size_t)(k0 + kk) * N]);
    }
    if (k0 + 32 < K) {  // L2/WGP prefetch of the next K-slab
      __builtin_prefetch(aptr + k0 + 32, 0, 1);
      __builtin_prefetch(wptr + (size_t)(k0 + 32 + bkk0) * N, 0, 1);
    }
    __syncthreads();

    Frag af[2], bf[4];
    #pragma unroll
    for (int mi = 0; mi < 2; ++mi) {
      int rl = wm * 32 + mi * 16 + lm;
      #pragma unroll
      for (int v = 0; v < 8; ++v) {
        int kp = ((v < 4) ? 0 : 16) + (v & 3) * 2 + hx * 8;  // A-frag K pairs
        af[mi].u[v] = *(const uint32*)&As[rl][kp];
      }
    }
    #pragma unroll
    for (int ni = 0; ni < 4; ++ni) {
      int cl = wn * 64 + ni * 16 + lm;
      #pragma unroll
      for (int v = 0; v < 8; ++v) {
        int kp = ((v < 4) ? 0 : 16) + (v & 3) * 2 + hx * 8;  // B-frag K pairs
        bf[ni].u[v] = *(const uint32*)&Bs[cl][kp];
      }
    }
    #pragma unroll
    for (int mi = 0; mi < 2; ++mi)
      #pragma unroll
      for (int ni = 0; ni < 4; ++ni)
        acc[mi][ni] = wmma_bf16(af[mi], bf[ni], acc[mi][ni]);
    __syncthreads();
  }

  #pragma unroll
  for (int mi = 0; mi < 2; ++mi)
    #pragma unroll
    for (int ni = 0; ni < 4; ++ni)
      #pragma unroll
      for (int r = 0; r < 8; ++r) {
        int grow = m0 + wm * 32 + mi * 16 + hx * 8 + r;
        int gcol = n0 + wn * 64 + ni * 16 + lm;
        float val = acc[mi][ni][r] + bias[gcol];
        size_t idx = (size_t)grow * N + gcol;
        if (EPI == 0) {
          outb[idx] = f2bf(val);
        } else if (EPI == 1) {
          float g = 0.5f * val * (1.0f + erff(val * 0.70710678118f));
          outb[idx] = f2bf(g);
        } else {
          outf[idx] = val + resid[idx];
        }
      }
}

// ---------------------------------------------------------------------------
// Attention key-block body. oacc[0..3] = O accumulator (d chunks of 16),
// oacc[4] = running softmax denominator via a "ones-column" WMMA: the row sum
// of P shares the exact rescaling recurrence of O, so P @ ones tracks l.
// ---------------------------------------------------------------------------
template <bool MASKED>
__device__ __forceinline__ void attn_block(
    int kbase, int q0, int lm, int hx,
    const ushort_t* __restrict__ kp, const ushort_t* __restrict__ vp,
    const Frag (&qf)[2], v8f (&oacc)[5], float (&mrow)[8],
    ushort_t (*pst)[34]) {
  const int C3 = 3072;
  float sv[2][8];
  float mb[8];
  #pragma unroll
  for (int r = 0; r < 8; ++r) mb[r] = -1e30f;

  #pragma unroll
  for (int sub = 0; sub < 2; ++sub) {
    Frag kf[2];
    int key = kbase + sub * 16 + lm;
    #pragma unroll
    for (int c = 0; c < 2; ++c)
      #pragma unroll
      for (int v = 0; v < 8; ++v) {
        int d = c * 32 + ((v < 4) ? 0 : 16) + (v & 3) * 2 + hx * 8;  // B-frag: K^T, two d per VGPR
        kf[c].u[v] = *(const uint32*)&kp[(size_t)key * C3 + d];
      }
    v8f s = {};
    s = wmma_bf16(qf[0], kf[0], s);
    s = wmma_bf16(qf[1], kf[1], s);
    #pragma unroll
    for (int r = 0; r < 8; ++r) {
      float x = s[r] * 0.125f;  // 1/sqrt(64)
      if (MASKED) {
        int row = q0 + hx * 8 + r;
        x = (key <= row) ? x : -1e30f;
      }
      sv[sub][r] = x;
      mb[r] = fmaxf(mb[r], x);
    }
  }
  #pragma unroll
  for (int r = 0; r < 8; ++r) {
    #pragma unroll
    for (int o = 1; o < 16; o <<= 1)
      mb[r] = fmaxf(mb[r], __shfl_xor(mb[r], o, 32));
    float mn = fmaxf(mrow[r], mb[r]);
    float corr = __expf(mrow[r] - mn);
    mrow[r] = mn;
    #pragma unroll
    for (int c = 0; c < 5; ++c) oacc[c][r] *= corr;
  }
  #pragma unroll
  for (int sub = 0; sub < 2; ++sub)
    #pragma unroll
    for (int r = 0; r < 8; ++r) {
      float e;
      if (MASKED)
        e = (sv[sub][r] <= -1e29f) ? 0.f : __expf(sv[sub][r] - mrow[r]);
      else
        e = __expf(sv[sub][r] - mrow[r]);
      pst[hx * 8 + r][sub * 16 + lm] = f2bf(e);
    }
  // Same-wave LDS visibility (waves have divergent trip counts -> no
  // __syncthreads). LDS is in-order per wave; just wait for DS writes.
  __asm__ volatile("s_wait_dscnt 0" ::: "memory");
  Frag pf;  // P as 16x32 A-frag
  #pragma unroll
  for (int v = 0; v < 8; ++v) {
    int kx = ((v < 4) ? 0 : 16) + (v & 3) * 2 + hx * 8;
    pf.u[v] = *(const uint32*)&pst[lm][kx];
  }
  #pragma unroll
  for (int c = 0; c < 4; ++c) {
    Frag vf;  // V block as 32x16 B-frag (keys = reduction dim)
    #pragma unroll
    for (int v = 0; v < 8; ++v) {
      int key = kbase + ((v < 4) ? 0 : 16) + (v & 3) * 2 + hx * 8;
      int d = c * 16 + lm;
      uint32 lo = vp[(size_t)key * C3 + d];
      uint32 hi = vp[(size_t)(key + 1) * C3 + d];
      vf.u[v] = lo | (hi << 16);
    }
    oacc[c] = wmma_bf16(pf, vf, oacc[c]);
  }
  Frag ones;  // ones column: row-sum of P (softmax denominator)
  #pragma unroll
  for (int v = 0; v < 8; ++v) ones.u[v] = 0x3F803F80u;  // bf16 1.0 x2
  oacc[4] = wmma_bf16(pf, ones, oacc[4]);
  __asm__ volatile("" ::: "memory");
}

// ---------------------------------------------------------------------------
// Causal attention, flash-style online softmax. Grid (T/64, H, B), 128 thr.
// Wave w owns 16 query rows. qkv bf16 [B,T,3072]; head h at col h*64 (+1024 K,
// +2048 V). Output y bf16 [B,T,1024]. Key blocks split into unmasked
// (below-diagonal) and masked (diagonal-straddling) paths.
// ---------------------------------------------------------------------------
__global__ __launch_bounds__(128) void attn_kernel(
    const ushort_t* __restrict__ qkv, ushort_t* __restrict__ y) {
  const int T = 2048, C3 = 3072;
  int tid = threadIdx.x;
  int l = tid & 31, w = tid >> 5;
  int lm = l & 15, hx = l >> 4;
  int h = blockIdx.y, b = blockIdx.z;
  int q0 = blockIdx.x * 64 + w * 16;

  const ushort_t* qp = qkv + (size_t)b * T * C3 + h * 64;
  const ushort_t* kp = qp + 1024;
  const ushort_t* vp = qp + 2048;

  __shared__ ushort_t Pst[4][16][34];  // per-wave P staging (C-layout -> A)

  Frag qf[2];  // Q rows q0..q0+15, d = 0..63 as two 16x32 A-frags
  #pragma unroll
  for (int c = 0; c < 2; ++c)
    #pragma unroll
    for (int v = 0; v < 8; ++v) {
      int d = c * 32 + ((v < 4) ? 0 : 16) + (v & 3) * 2 + hx * 8;
      qf[c].u[v] = *(const uint32*)&qp[(size_t)(q0 + lm) * C3 + d];
    }

  v8f oacc[5] = {};
  float mrow[8];
  #pragma unroll
  for (int r = 0; r < 8; ++r) mrow[r] = -1e30f;

  int nfull = (q0 + 1) >> 5;   // blocks with kbase+31 <= q0: no masking
  int kbmax = (q0 + 15) >> 5;  // last block touching row q0+15
  for (int kb = 0; kb < nfull; ++kb)
    attn_block<false>(kb << 5, q0, lm, hx, kp, vp, qf, oacc, mrow, Pst[w]);
  for (int kb = nfull; kb <= kbmax; ++kb)
    attn_block<true>(kb << 5, q0, lm, hx, kp, vp, qf, oacc, mrow, Pst[w]);

  #pragma unroll
  for (int r = 0; r < 8; ++r) {
    float inv = 1.0f / oacc[4][r];  // denominator, replicated across N lanes
    int row = q0 + hx * 8 + r;
    #pragma unroll
    for (int c = 0; c < 4; ++c)
      y[((size_t)b * T + row) * 1024 + h * 64 + c * 16 + lm] =
          f2bf(oacc[c][r] * inv);
  }
}

// ---------------------------------------------------------------------------
extern "C" void kernel_launch(void* const* d_in, const int* in_sizes, int n_in,
                              void* d_out, int out_size, void* d_ws,
                              size_t ws_size, hipStream_t stream) {
  (void)in_sizes; (void)n_in; (void)out_size; (void)ws_size;
  const float* x      = (const float*)d_in[0];
  const float* ln1_w  = (const float*)d_in[1];
  const float* ln1_b  = (const float*)d_in[2];
  const float* ln2_w  = (const float*)d_in[3];
  const float* ln2_b  = (const float*)d_in[4];
  const float* w_attn = (const float*)d_in[5];
  const float* b_attn = (const float*)d_in[6];
  const float* w_proj = (const float*)d_in[7];
  const float* b_proj = (const float*)d_in[8];
  const float* w_fc   = (const float*)d_in[9];
  const float* b_fc   = (const float*)d_in[10];
  const float* w_out  = (const float*)d_in[11];
  const float* b_out  = (const float*)d_in[12];

  char* ws = (char*)d_ws;
  ushort_t* ln1buf = (ushort_t*)(ws + 0);          // 4096*1024 bf16 =  8 MB
  ushort_t* qkvbuf = (ushort_t*)(ws + 8388608);    // 4096*3072 bf16 = 24 MB
  ushort_t* ybuf   = (ushort_t*)(ws + 33554432);   // 4096*1024 bf16 =  8 MB
  float*    x2buf  = (float*)(ws + 41943040);      // 4096*1024 f32  = 16 MB
  ushort_t* h2buf  = (ushort_t*)(ws + 58720256);   // 4096*1024 bf16 =  8 MB
  ushort_t* actbuf = (ushort_t*)(ws + 67108864);   // 4096*4096 bf16 = 32 MB

  const int M = 4096;  // B*T

  ln_kernel<<<M, 256, 0, stream>>>(x, ln1_w, ln1_b, ln1buf);
  gemm_bf16_kernel<0><<<dim3(24, 32), 256, 0, stream>>>(
      ln1buf, w_attn, b_attn, nullptr, qkvbuf, nullptr, M, 3072, 1024);
  attn_kernel<<<dim3(32, 16, 2), 128, 0, stream>>>(qkvbuf, ybuf);
  gemm_bf16_kernel<2><<<dim3(8, 32), 256, 0, stream>>>(
      ybuf, w_proj, b_proj, x, nullptr, x2buf, M, 1024, 1024);
  ln_kernel<<<M, 256, 0, stream>>>(x2buf, ln2_w, ln2_b, h2buf);
  gemm_bf16_kernel<1><<<dim3(32, 32), 256, 0, stream>>>(
      h2buf, w_fc, b_fc, nullptr, actbuf, nullptr, M, 4096, 1024);
  gemm_bf16_kernel<2><<<dim3(8, 32), 256, 0, stream>>>(
      actbuf, w_out, b_out, x2buf, nullptr, (float*)d_out, M, 1024, 4096);
}
